// AIA_7937099563025
// MI455X (gfx1250) — matrix operand order repaired and checked by reading, not compiled, not run on hardware
//
#include <hip/hip_runtime.h>
#include <math.h>

// ---------------------------------------------------------------------------
// MI455X (gfx1250) implementation.
//  - N-sized stages (PE/scatter, epilogue): HBM-bound, wave-per-point, f32
//    global atomics. ~300 MB x3 -> ~40 us at 23.3 TB/s; VALU negligible.
//  - Attention core (~21 GFLOP): f16 WMMA (v_wmma_f32_16x16x32_f16), f32
//    accum. Every operand <= 40 MB -> L2-resident (192 MB), so a
//    direct-from-global WMMA kernel is bandwidth-safe. All B operands are
//    pre-transposed to N x K so each B fragment is two b128 loads per lane.
//    Each wave owns a 16x64 strip: 4 independent accumulators per K-step.
//  - FPS: inherently sequential -> one 1024-thread workgroup, dmin in LDS.
// ---------------------------------------------------------------------------

typedef _Float16 half_t;
typedef __attribute__((ext_vector_type(16))) _Float16 v16h;
typedef __attribute__((ext_vector_type(8)))  _Float16 v8h;
typedef __attribute__((ext_vector_type(8)))  float    v8f;

static constexpr int kN   = 150000;
static constexpr int kF   = 512;
static constexpr int kS   = 5000;
static constexpr int kK   = 512;   // number of FPS keys
static constexpr int kH   = 4;
static constexpr int kDH  = kF / kH;   // 128
static constexpr int kOut = 20;
static constexpr int kL   = 2;

// ---------------- wave helpers (wave32) ----------------
__device__ inline float wave_sum(float v) {
  #pragma unroll
  for (int m = 16; m >= 1; m >>= 1) v += __shfl_xor(v, m, 32);
  return v;
}
__device__ inline float wave_max(float v) {
  #pragma unroll
  for (int m = 16; m >= 1; m >>= 1) v = fmaxf(v, __shfl_xor(v, m, 32));
  return v;
}
__device__ inline v16h cat8(v8h lo, v8h hi) {
  return __builtin_shufflevector(lo, hi, 0, 1, 2, 3, 4, 5, 6, 7, 8, 9, 10, 11,
                                 12, 13, 14, 15);
}

// ---------------- utility kernels ----------------
__global__ void k_zero(float* __restrict__ p, long n) {
  for (long i = (long)blockIdx.x * blockDim.x + threadIdx.x; i < n;
       i += (long)gridDim.x * blockDim.x)
    p[i] = 0.0f;
}

__global__ void k_f2h(const float* __restrict__ s, half_t* __restrict__ d, long n) {
  for (long i = (long)blockIdx.x * blockDim.x + threadIdx.x; i < n;
       i += (long)gridDim.x * blockDim.x)
    d[i] = (half_t)s[i];
}

// transpose-convert: in (R x C, f32) -> out (C x R, f16); R, C multiples of 16
__global__ __launch_bounds__(256) void k_f2h_t(const float* __restrict__ s,
                                               half_t* __restrict__ d, int R, int C) {
  __shared__ float tile[16][17];
  int tx = threadIdx.x & 15, ty = threadIdx.x >> 4;
  int r = blockIdx.y * 16 + ty, c = blockIdx.x * 16 + tx;
  tile[ty][tx] = s[(size_t)r * C + c];
  __syncthreads();
  int oc = blockIdx.x * 16 + ty, orr = blockIdx.y * 16 + tx;
  d[(size_t)oc * R + orr] = (half_t)tile[tx][ty];
}

// ---------------- coords min/max ----------------
__global__ __launch_bounds__(256) void k_minmax_part(const float* __restrict__ coords,
                                                     float* __restrict__ part, int Np) {
  __shared__ float smn[3][256];
  __shared__ float smx[3][256];
  int t = threadIdx.x;
  float mn[3] = {1e30f, 1e30f, 1e30f};
  float mx[3] = {-1e30f, -1e30f, -1e30f};
  for (size_t p = (size_t)blockIdx.x * 256 + t; p < (size_t)Np;
       p += (size_t)gridDim.x * 256) {
    #pragma unroll
    for (int d = 0; d < 3; d++) {
      float v = coords[p * 3 + d];
      mn[d] = fminf(mn[d], v);
      mx[d] = fmaxf(mx[d], v);
    }
  }
  #pragma unroll
  for (int d = 0; d < 3; d++) { smn[d][t] = mn[d]; smx[d][t] = mx[d]; }
  __syncthreads();
  for (int off = 128; off >= 1; off >>= 1) {
    if (t < off) {
      #pragma unroll
      for (int d = 0; d < 3; d++) {
        smn[d][t] = fminf(smn[d][t], smn[d][t + off]);
        smx[d][t] = fmaxf(smx[d][t], smx[d][t + off]);
      }
    }
    __syncthreads();
  }
  if (t == 0) {
    #pragma unroll
    for (int d = 0; d < 3; d++) {
      part[blockIdx.x * 6 + d]     = smn[d][0];
      part[blockIdx.x * 6 + 3 + d] = smx[d][0];
    }
  }
}

__global__ __launch_bounds__(256) void k_minmax_final(const float* __restrict__ part,
                                                      float* __restrict__ mnmx, int nb) {
  __shared__ float smn[3][256];
  __shared__ float smx[3][256];
  int t = threadIdx.x;
  float mn[3] = {1e30f, 1e30f, 1e30f};
  float mx[3] = {-1e30f, -1e30f, -1e30f};
  for (int i = t; i < nb; i += 256) {
    #pragma unroll
    for (int d = 0; d < 3; d++) {
      mn[d] = fminf(mn[d], part[i * 6 + d]);
      mx[d] = fmaxf(mx[d], part[i * 6 + 3 + d]);
    }
  }
  #pragma unroll
  for (int d = 0; d < 3; d++) { smn[d][t] = mn[d]; smx[d][t] = mx[d]; }
  __syncthreads();
  for (int off = 128; off >= 1; off >>= 1) {
    if (t < off) {
      #pragma unroll
      for (int d = 0; d < 3; d++) {
        smn[d][t] = fminf(smn[d][t], smn[d][t + off]);
        smx[d][t] = fmaxf(smx[d][t], smx[d][t + off]);
      }
    }
    __syncthreads();
  }
  if (t == 0) {
    #pragma unroll
    for (int d = 0; d < 3; d++) { mnmx[d] = smn[d][0]; mnmx[3 + d] = smx[d][0]; }
  }
}

// ---------------- Fourier PE + scatter-mean accumulation (wave per point) ----
__global__ __launch_bounds__(256) void k_scatter(const float* __restrict__ pf,
                                                 const float* __restrict__ coords,
                                                 const float* __restrict__ B,
                                                 const int* __restrict__ seg,
                                                 const float* __restrict__ mnmx,
                                                 float* __restrict__ fsum,
                                                 float* __restrict__ psum,
                                                 float* __restrict__ cnt, int Np) {
  int lane = threadIdx.x & 31, wv = threadIdx.x >> 5;
  long p = (long)blockIdx.x * 8 + wv;
  if (p >= Np) return;
  int s = seg[p];
  float cn[3];
  #pragma unroll
  for (int d = 0; d < 3; d++) {
    float mn = mnmx[d], mx = mnmx[3 + d];
    cn[d] = (coords[p * 3 + d] - mn) / (mx - mn + 1e-6f);
  }
  float* fs = fsum + (size_t)s * kF;
  float* ps = psum + (size_t)s * kF;
  const float* pr = pf + (size_t)p * kF;
  #pragma unroll
  for (int j = 0; j < 16; j++) {
    int c = lane * 16 + j;
    atomicAdd(&fs[c], pr[c]);
    int freq = (c < 256) ? c : (c - 256);
    float proj = 6.28318530717958647692f *
                 (cn[0] * B[freq] + cn[1] * B[256 + freq] + cn[2] * B[512 + freq]);
    float v = (c < 256) ? sinf(proj) : cosf(proj);
    atomicAdd(&ps[c], v);
  }
  if (lane == 0) atomicAdd(&cnt[s], 1.0f);
}

// sums -> means; also q0 = seg_feat
__global__ void k_seg_finalize(float* __restrict__ fsum, float* __restrict__ psum,
                               const float* __restrict__ cnt, float* __restrict__ q,
                               long n) {
  for (long i = (long)blockIdx.x * blockDim.x + threadIdx.x; i < n;
       i += (long)gridDim.x * blockDim.x) {
    int s = (int)(i >> 9);
    float c = fmaxf(cnt[s], 1.0f);
    float f = fsum[i] / c;
    fsum[i] = f;
    q[i] = f;
    psum[i] = psum[i] / c;
  }
}

// ---------------- FPS (sequential, single workgroup) ----------------
__global__ __launch_bounds__(1024) void k_fps(const float* __restrict__ segpos,
                                              int* __restrict__ fps_idx) {
  __shared__ float dmin[kS];
  __shared__ __align__(16) float xcur[kF];
  __shared__ float rv[1024];
  __shared__ int   ri[1024];
  __shared__ int   cur;
  int t = threadIdx.x;
  for (int s = t; s < kS; s += 1024) dmin[s] = 1e10f;
  if (t == 0) cur = 0;
  __syncthreads();
  const float4* xc4 = (const float4*)xcur;
  for (int step = 0; step < kK; ++step) {
    int c = cur;
    if (t == 0) fps_idx[step] = c;
    if (t < kF) xcur[t] = segpos[(size_t)c * kF + t];
    __syncthreads();
    float bv = -1.0f;
    int bi = 0x7fffffff;
    for (int s = t; s < kS; s += 1024) {
      const float4* row4 = (const float4*)(segpos + (size_t)s * kF);
      float d = 0.0f;
      #pragma unroll 4
      for (int j = 0; j < kF / 4; j++) {
        float4 r4 = row4[j];
        float4 x4 = xc4[j];
        float d0 = r4.x - x4.x, d1 = r4.y - x4.y;
        float d2 = r4.z - x4.z, d3 = r4.w - x4.w;
        d = fmaf(d0, d0, d); d = fmaf(d1, d1, d);
        d = fmaf(d2, d2, d); d = fmaf(d3, d3, d);
      }
      float dm = fminf(dmin[s], d);
      dmin[s] = dm;
      if (dm > bv) { bv = dm; bi = s; }
    }
    rv[t] = bv; ri[t] = bi;
    __syncthreads();
    for (int off = 512; off >= 1; off >>= 1) {
      if (t < off) {
        float ov = rv[t + off]; int oi = ri[t + off];
        if (ov > rv[t] || (ov == rv[t] && oi < ri[t])) { rv[t] = ov; ri[t] = oi; }
      }
      __syncthreads();
    }
    if (t == 0) cur = ri[0];
    __syncthreads();
  }
}

// ---------------- gather keys ----------------
__global__ __launch_bounds__(256) void k_keys(const float* __restrict__ segfeat,
                                              const float* __restrict__ segpos,
                                              const int* __restrict__ idx,
                                              half_t* __restrict__ keysum16,
                                              half_t* __restrict__ keyfeat16) {
  int r = blockIdx.x;
  int s = idx[r];
  for (int c = threadIdx.x; c < kF; c += 256) {
    float f = segfeat[(size_t)s * kF + c];
    float p = segpos[(size_t)s * kF + c];
    keyfeat16[(size_t)r * kF + c] = (half_t)f;
    keysum16[(size_t)r * kF + c]  = (half_t)(f + p);
  }
}

// ---------------- LayerNorm + add pos, emit f16 (wave per row) ----------------
__global__ __launch_bounds__(256) void k_ln_pe(const float* __restrict__ q,
                                               const float* __restrict__ g,
                                               const float* __restrict__ b,
                                               const float* __restrict__ pos,
                                               half_t* __restrict__ X, int rows) {
  int lane = threadIdx.x & 31, wv = threadIdx.x >> 5;
  long r = (long)blockIdx.x * 8 + wv;
  if (r >= rows) return;
  const float* row = q + (size_t)r * kF;
  const float* prow = pos + (size_t)r * kF;
  half_t* xr = X + (size_t)r * kF;
  float s = 0.0f, s2 = 0.0f;
  for (int c = lane; c < kF; c += 32) { float v = row[c]; s += v; s2 = fmaf(v, v, s2); }
  s = wave_sum(s); s2 = wave_sum(s2);
  float m = s * (1.0f / kF);
  float var = s2 * (1.0f / kF) - m * m;
  float inv = rsqrtf(var + 1e-5f);
  for (int c = lane; c < kF; c += 32)
    xr[c] = (half_t)(((row[c] - m) * inv) * g[c] + b[c] + prow[c]);
}

// ---------------- WMMA GEMM: C[,h] = A[,h] x Bt[,h]^T (+Cinit +bias) ---------
// A:  M x Kd row-major (lda).   Bt: N x Kd row-major (ldb) == B transposed.
// Wave computes a 16x64 strip: one A fragment, 4 independent accumulators.
// N must be a multiple of 64; Kd a multiple of 32. OOB M rows are clamped
// (they only feed C rows that are never stored).
__global__ __launch_bounds__(256) void k_gemm4(
    const half_t* __restrict__ A, int lda, long headA,
    const half_t* __restrict__ Bt, int ldb, long headB,
    float* __restrict__ C, int ldc, long headC,
    const float* __restrict__ Cinit, long headCi,
    const float* __restrict__ bias,
    int M, int N, int Kd) {
  int h = blockIdx.y;
  A  += (size_t)h * headA;
  Bt += (size_t)h * headB;
  C  += (size_t)h * headC;
  if (Cinit) Cinit += (size_t)h * headCi;

  int lane = threadIdx.x & 31, wv = threadIdx.x >> 5;
  int ntN = N >> 6;
  int ntM = (M + 15) >> 4;
  long tile = (long)blockIdx.x * 8 + wv;
  if (tile >= (long)ntM * ntN) return;
  int tm = (int)(tile / ntN), tn = (int)(tile % ntN);
  int row0 = tm << 4, col0 = tn << 6;
  int l15 = lane & 15, lhi = lane >> 4;

  // A fragment: lane<16 -> K kk+0..15; lane>=16 -> kk+16..31
  int arow = row0 + l15;
  if (arow > M - 1) arow = M - 1;
  const half_t* ap = A + (size_t)arow * lda + lhi * 16;
  // B fragment (from Bt): lane col = l15; lane<16 -> K kk+0..15; else kk+16..31
  const half_t* bp[4];
  #pragma unroll
  for (int t = 0; t < 4; t++)
    bp[t] = Bt + (size_t)(col0 + t * 16 + l15) * ldb + lhi * 16;

  v8f acc[4];
  #pragma unroll
  for (int t = 0; t < 4; t++) {
    int ccol = col0 + t * 16 + l15;
    float bb = bias ? bias[ccol] : 0.0f;
    #pragma unroll
    for (int r = 0; r < 8; r++) {
      int crow = row0 + r + (lhi << 3);
      if (crow > M - 1) crow = M - 1;
      float ci = Cinit ? Cinit[(size_t)crow * ldc + ccol] : 0.0f;
      acc[t][r] = ci + bb;
    }
  }

  for (int kk = 0; kk < Kd; kk += 32) {
    v8h alo = *(const v8h*)(ap + kk);
    v8h ahi = *(const v8h*)(ap + kk + 8);
    v16h a = cat8(alo, ahi);
    __builtin_prefetch(ap + kk + 32, 0, 0);  // stream the A panel
    #pragma unroll
    for (int t = 0; t < 4; t++) {
      v8h blo = *(const v8h*)(bp[t] + kk);
      v8h bhi = *(const v8h*)(bp[t] + kk + 8);
      v16h b = cat8(blo, bhi);
      acc[t] = __builtin_amdgcn_wmma_f32_16x16x32_f16(false, a, false, b, (short)0,
                                                      acc[t], false, false);
    }
  }

  #pragma unroll
  for (int t = 0; t < 4; t++) {
    int ccol = col0 + t * 16 + l15;
    #pragma unroll
    for (int r = 0; r < 8; r++) {
      int crow = row0 + r + (lhi << 3);
      if (crow < M) C[(size_t)crow * ldc + ccol] = acc[t][r];
    }
  }
}

// ---------------- softmax over K (wave per row), emit f16 ----------------
__global__ __launch_bounds__(256) void k_softmax(const float* __restrict__ S_,
                                                 half_t* __restrict__ att, int rows,
                                                 float scale) {
  int lane = threadIdx.x & 31, wv = threadIdx.x >> 5;
  long r = (long)blockIdx.x * 8 + wv;
  if (r >= rows) return;
  const float* row = S_ + r * (long)kK;
  half_t* orow = att + r * (long)kK;
  float lmax = -1e30f;
  for (int c = lane; c < kK; c += 32) lmax = fmaxf(lmax, row[c] * scale);
  lmax = wave_max(lmax);
  float lsum = 0.0f;
  for (int c = lane; c < kK; c += 32) lsum += expf(row[c] * scale - lmax);
  lsum = wave_sum(lsum);
  float inv = 1.0f / lsum;
  for (int c = lane; c < kK; c += 32)
    orow[c] = (half_t)(expf(row[c] * scale - lmax) * inv);
}

// ---------------- fused epilogue (wave per point) ----------------
__global__ __launch_bounds__(256) void k_final(const float* __restrict__ q,
                                               const float* __restrict__ pf,
                                               const int* __restrict__ seg,
                                               const float* __restrict__ g1,
                                               const float* __restrict__ b1,
                                               const float* __restrict__ w1,
                                               const float* __restrict__ b1s,
                                               const float* __restrict__ cw,
                                               const float* __restrict__ cb,
                                               float* __restrict__ out, int Np) {
  int lane = threadIdx.x & 31, wv = threadIdx.x >> 5;
  long p = (long)blockIdx.x * 8 + wv;
  if (p >= Np) return;
  int s = seg[p];
  const float4* fr4 = (const float4*)(q + (size_t)s * kF);
  const float4* pr4 = (const float4*)(pf + (size_t)p * kF);
  float f[16], pv[16], dabs[16];
  #pragma unroll
  for (int g = 0; g < 4; g++) {
    float4 a4 = fr4[lane * 4 + g];
    float4 b4 = pr4[lane * 4 + g];
    f[g * 4 + 0] = a4.x; f[g * 4 + 1] = a4.y; f[g * 4 + 2] = a4.z; f[g * 4 + 3] = a4.w;
    pv[g * 4 + 0] = b4.x; pv[g * 4 + 1] = b4.y; pv[g * 4 + 2] = b4.z; pv[g * 4 + 3] = b4.w;
  }
  float sm = 0.0f, sq = 0.0f;
  #pragma unroll
  for (int j = 0; j < 16; j++) {
    float d = fabsf(f[j] - pv[j]);
    dabs[j] = d;
    sm += d;
    sq = fmaf(d, d, sq);
  }
  sm = wave_sum(sm);
  sq = wave_sum(sq);
  float m = sm * (1.0f / kF);
  float var = sq * (1.0f / kF) - m * m;
  float inv = rsqrtf(var + 1e-5f);
  float wp = 0.0f;
  #pragma unroll
  for (int j = 0; j < 16; j++) {
    int c = lane * 16 + j;
    wp += ((dabs[j] - m) * inv * g1[c] + b1[c]) * w1[c];
  }
  wp = wave_sum(wp);
  float w = 1.0f / (1.0f + expf(-(wp + b1s[0])));
  float bl[16];
  #pragma unroll
  for (int j = 0; j < 16; j++) bl[j] = (1.0f - w) * pv[j] + w * f[j];
  float o[kOut];
  #pragma unroll
  for (int oo = 0; oo < kOut; oo++) o[oo] = 0.0f;
  #pragma unroll
  for (int j = 0; j < 16; j++) {
    int c = lane * 16 + j;
    const float* cwr = cw + (size_t)c * kOut;
    #pragma unroll
    for (int oo = 0; oo < kOut; oo++) o[oo] = fmaf(bl[j], cwr[oo], o[oo]);
  }
  #pragma unroll
  for (int oo = 0; oo < kOut; oo++) {
    float v = wave_sum(o[oo]);
    if (lane == 0) out[p * (long)kOut + oo] = v + cb[oo];
  }
}

// ---------------------------------------------------------------------------
extern "C" void kernel_launch(void* const* d_in, const int* in_sizes, int n_in,
                              void* d_out, int out_size, void* d_ws, size_t ws_size,
                              hipStream_t stream) {
  (void)in_sizes; (void)n_in; (void)out_size; (void)ws_size;
  const float* point_feats = (const float*)d_in[0];
  const float* coords      = (const float*)d_in[1];
  const float* fourier_B   = (const float*)d_in[2];
  const float* ln_g = (const float*)d_in[3];
  const float* ln_b = (const float*)d_in[4];
  const float* Wq = (const float*)d_in[5];
  const float* Wk = (const float*)d_in[6];
  const float* Wv = (const float*)d_in[7];
  const float* Wo = (const float*)d_in[8];
  const float* bq = (const float*)d_in[9];
  const float* bk = (const float*)d_in[10];
  const float* bv = (const float*)d_in[11];
  const float* bo = (const float*)d_in[12];
  const float* c1g  = (const float*)d_in[13];
  const float* c1b  = (const float*)d_in[14];
  const float* c1w  = (const float*)d_in[15];
  const float* c1bs = (const float*)d_in[16];
  const float* cw = (const float*)d_in[17];
  const float* cb = (const float*)d_in[18];
  const int* p2s = (const int*)d_in[19];
  float* out = (float*)d_out;

  // workspace carve-out
  char* ws = (char*)d_ws;
  size_t off = 0;
  auto take = [&](size_t bytes) -> char* {
    char* p = ws + off;
    off = (off + bytes + 255) & ~(size_t)255;
    return p;
  };
  float* part    = (float*)take(256 * 6 * 4);
  float* mnmx    = (float*)take(8 * 4);
  float* cnt     = (float*)take((size_t)kS * 4);
  float* segfeat = (float*)take((size_t)kS * kF * 4);
  float* segpos  = (float*)take((size_t)kS * kF * 4);
  float* qbuf    = (float*)take((size_t)kS * kF * 4);
  float* qh      = (float*)take((size_t)kS * kF * 4);
  float* kh      = (float*)take((size_t)kK * kF * 4);
  float* vh      = (float*)take((size_t)kK * kF * 4);
  float* scores  = (float*)take((size_t)kH * kS * kK * 4);
  float* Obuf    = (float*)take((size_t)kS * kF * 4);
  int*   fpsidx  = (int*)take((size_t)kK * 4);
  half_t* X16      = (half_t*)take((size_t)kS * kF * 2);
  half_t* W16T     = (half_t*)take((size_t)kF * kF * 2);   // transposed weights
  half_t* qh16     = (half_t*)take((size_t)kS * kF * 2);
  half_t* kh16     = (half_t*)take((size_t)kK * kF * 2);
  half_t* vhT16    = (half_t*)take((size_t)kK * kF * 2);   // transposed V
  half_t* att16    = (half_t*)take((size_t)kH * kS * kK * 2);
  half_t* O16      = (half_t*)take((size_t)kS * kF * 2);
  half_t* keysum16 = (half_t*)take((size_t)kK * kF * 2);
  half_t* keyfeat16= (half_t*)take((size_t)kK * kF * 2);

  auto gemm_grid = [](int M, int N) {
    long t = (long)((M + 15) / 16) * (N / 64);
    return (int)((t + 7) / 8);
  };
  const float scale = 0.08838834764831844f;  // 1/sqrt(dh=128)
  const dim3 t16x16(32, 32);                 // 512x512 transpose grid

  // 1) zero accumulators
  k_zero<<<1024, 256, 0, stream>>>(cnt, (long)kS);
  k_zero<<<2048, 256, 0, stream>>>(segfeat, (long)kS * kF);
  k_zero<<<2048, 256, 0, stream>>>(segpos, (long)kS * kF);

  // 2) coords min/max
  k_minmax_part<<<256, 256, 0, stream>>>(coords, part, kN);
  k_minmax_final<<<1, 256, 0, stream>>>(part, mnmx, 256);

  // 3) PE + scatter-mean sums (wave per point)
  k_scatter<<<(kN + 7) / 8, 256, 0, stream>>>(point_feats, coords, fourier_B, p2s,
                                              mnmx, segfeat, segpos, cnt, kN);

  // 4) sums -> means, q0 = seg_feat
  k_seg_finalize<<<2048, 256, 0, stream>>>(segfeat, segpos, cnt, qbuf, (long)kS * kF);

  // 5) furthest point sampling (sequential)
  k_fps<<<1, 1024, 0, stream>>>(segpos, fpsidx);

  // 6) gather keys, pre-add key_pos, convert to f16
  k_keys<<<kK, 256, 0, stream>>>(segfeat, segpos, fpsidx, keysum16, keyfeat16);

  // 7) two pre-norm cross-attention layers
  for (int l = 0; l < kL; l++) {
    // X16 = f16(LN(q)*g+b + seg_pos)
    k_ln_pe<<<(kS + 7) / 8, 256, 0, stream>>>(qbuf, ln_g + (size_t)l * kF,
                                              ln_b + (size_t)l * kF, segpos, X16, kS);
    // qh = X @ Wq + bq   (Wq pre-transposed to N x K)
    k_f2h_t<<<t16x16, 256, 0, stream>>>(Wq + (size_t)l * kF * kF, W16T, kF, kF);
    k_gemm4<<<dim3(gemm_grid(kS, kF), 1), 256, 0, stream>>>(
        X16, kF, 0, W16T, kF, 0, qh, kF, 0, nullptr, 0, bq + (size_t)l * kF,
        kS, kF, kF);
    k_f2h<<<2048, 256, 0, stream>>>(qh, qh16, (long)kS * kF);
    // kh = (key_feat+key_pos) @ Wk + bk
    k_f2h_t<<<t16x16, 256, 0, stream>>>(Wk + (size_t)l * kF * kF, W16T, kF, kF);
    k_gemm4<<<dim3(gemm_grid(kK, kF), 1), 256, 0, stream>>>(
        keysum16, kF, 0, W16T, kF, 0, kh, kF, 0, nullptr, 0, bk + (size_t)l * kF,
        kK, kF, kF);
    k_f2h<<<1024, 256, 0, stream>>>(kh, kh16, (long)kK * kF);
    // vh = key_feat @ Wv + bv; then transpose-convert vh -> vhT16 (F x K)
    k_f2h_t<<<t16x16, 256, 0, stream>>>(Wv + (size_t)l * kF * kF, W16T, kF, kF);
    k_gemm4<<<dim3(gemm_grid(kK, kF), 1), 256, 0, stream>>>(
        keyfeat16, kF, 0, W16T, kF, 0, vh, kF, 0, nullptr, 0, bv + (size_t)l * kF,
        kK, kF, kF);
    k_f2h_t<<<t16x16, 256, 0, stream>>>(vh, vhT16, kK, kF);
    // scores[h] = qh[:,h] @ kh[:,h]^T  (kh16 rows are already N x K layout)
    k_gemm4<<<dim3(gemm_grid(kS, kK), kH), 256, 0, stream>>>(
        qh16, kF, (long)kDH, kh16, kF, (long)kDH, scores, kK, (long)kS * kK,
        nullptr, 0, nullptr, kS, kK, kDH);
    // softmax (with 1/sqrt(dh) scale) -> f16
    k_softmax<<<(kH * kS + 7) / 8, 256, 0, stream>>>(scores, att16, kH * kS, scale);
    // O[:, h*dh:...] = att[h] @ vh[:, h*dh:...]  (B from vhT16, rows = out col)
    k_gemm4<<<dim3(gemm_grid(kS, kDH), kH), 256, 0, stream>>>(
        att16, kK, (long)kS * kK, vhT16, kK, (long)kDH * kK, Obuf, kF, (long)kDH,
        nullptr, 0, nullptr, kS, kDH, kK);
    k_f2h<<<2048, 256, 0, stream>>>(Obuf, O16, (long)kS * kF);
    // q = q + O @ Wo + bo  (residual+bias folded into WMMA C operand)
    k_f2h_t<<<t16x16, 256, 0, stream>>>(Wo + (size_t)l * kF * kF, W16T, kF, kF);
    k_gemm4<<<dim3(gemm_grid(kS, kF), 1), 256, 0, stream>>>(
        O16, kF, 0, W16T, kF, 0, qbuf, kF, 0, qbuf, 0, bo + (size_t)l * kF,
        kS, kF, kF);
  }

  // 8) gather-back + gated blend + classifier (wave per point)
  k_final<<<(kN + 7) / 8, 256, 0, stream>>>(qbuf, point_feats, p2s, c1g, c1b, c1w,
                                            c1bs, cw, cb, out, kN);
}
